// RNN_48481590837327
// MI455X (gfx1250) — matrix operand order repaired and checked by reading, not compiled, not run on hardware
//
#include <hip/hip_runtime.h>
#include <hip/hip_bf16.h>
#include <stdint.h>

#define B_ 32
#define E_ 128
#define H_ 128
#define V_ 50257
#define T_ 128

typedef __attribute__((ext_vector_type(16))) __bf16 v16bf;
typedef __attribute__((ext_vector_type(8)))  float  v8f;

struct __attribute__((aligned(16))) Q128 { unsigned int x, y, z, w; };
union Frag { Q128 q[2]; v16bf v; };

__device__ __forceinline__ unsigned short f2bf(float f) {
    unsigned int u = __float_as_uint(f);
    u += 0x7FFFu + ((u >> 16) & 1u);   // round-to-nearest-even
    return (unsigned short)(u >> 16);
}

__device__ __forceinline__ float sigm(float x) { return 1.0f / (1.0f + __expf(-x)); }

// ---------------- one-time: W_lin f32 -> bf16 (stays L2 resident, 12.9 MB) ----
__global__ void k_f32_to_bf16(const float* __restrict__ in,
                              unsigned short* __restrict__ out, int n) {
    int i = blockIdx.x * blockDim.x + threadIdx.x;
    if (i < n) out[i] = f2bf(in[i]);
}

// ---------------- per step: LSTM cell (fp32 recurrence, emits h in bf16) -----
__global__ __launch_bounds__(256)
void k_lstm_cell(const float* __restrict__ x, const float* __restrict__ hprev,
                 float* __restrict__ c, float* __restrict__ hcur,
                 unsigned short* __restrict__ hbf,
                 const float* __restrict__ Wih, const float* __restrict__ Whh,
                 const float* __restrict__ bih, const float* __restrict__ bhh) {
    int gid = blockIdx.x * blockDim.x + threadIdx.x;   // 0..4095
    int b = gid >> 7;
    int j = gid & 127;
    const float* xr = x + b * E_;
    const float* hr = hprev + b * H_;
    float gi = bih[j]          + bhh[j];
    float gf = bih[j + H_]     + bhh[j + H_];
    float gg = bih[j + 2 * H_] + bhh[j + 2 * H_];
    float go = bih[j + 3 * H_] + bhh[j + 3 * H_];
    const float* wi = Wih + (size_t)j * E_;
    const float* wh = Whh + (size_t)j * H_;
    for (int k = 0; k < E_; ++k) {
        float xv = xr[k];
        gi = fmaf(xv, wi[k],             gi);
        gf = fmaf(xv, wi[(size_t)H_ * E_ + k],     gf);
        gg = fmaf(xv, wi[(size_t)2 * H_ * E_ + k], gg);
        go = fmaf(xv, wi[(size_t)3 * H_ * E_ + k], go);
    }
    for (int k = 0; k < H_; ++k) {
        float hv = hr[k];
        gi = fmaf(hv, wh[k],             gi);
        gf = fmaf(hv, wh[(size_t)H_ * H_ + k],     gf);
        gg = fmaf(hv, wh[(size_t)2 * H_ * H_ + k], gg);
        go = fmaf(hv, wh[(size_t)3 * H_ * H_ + k], go);
    }
    float cv = sigm(gf) * c[gid] + sigm(gi) * tanhf(gg);
    c[gid] = cv;
    float hv = sigm(go) * tanhf(cv);
    hcur[gid] = hv;
    hbf[gid]  = f2bf(hv);
}

// ---------------- per step: logits GEMM via v_wmma_f32_16x16x32_bf16 ---------
// D[32, V] = h_bf16[32,128] x W_lin_bf16[V,128]^T + b_lin
// wave: 32(M) x 16(N) tile (2 accumulators); 8 waves/block -> 128 vocab cols.
__global__ __launch_bounds__(256)
void k_logits_wmma(const unsigned short* __restrict__ hbf,
                   const unsigned short* __restrict__ wbf,
                   const float* __restrict__ blin,
                   float* __restrict__ logits) {
    const int lane = threadIdx.x & 31;
    const int wave = threadIdx.x >> 5;
    const int col  = lane & 15;
    const int half = lane >> 4;
    const int v0   = blockIdx.x * 128 + wave * 16;
    const int vc   = v0 + col;
    const int vcl  = vc < V_ ? vc : (V_ - 1);

    // A layout (16-bit 16x32): lanes 0-15 row=lane K {0..7,16..23}; lanes 16-31 K {8..15,24..31}
    const unsigned short* a0_base = hbf + (size_t)col * H_ + half * 8;
    const unsigned short* a1_base = a0_base + 16 * H_;
    // B layout (16-bit 32x16): lane column = lane&15; lanes 0-15 K {0..7,16..23}; lanes 16-31 K {8..15,24..31}
    const unsigned short* b_base  = wbf + (size_t)vcl * H_ + half * 8;

    v8f acc0 = {0.f, 0.f, 0.f, 0.f, 0.f, 0.f, 0.f, 0.f};
    v8f acc1 = acc0;

#pragma unroll
    for (int kk = 0; kk < 4; ++kk) {
        const int kb = kk * 32;
        Frag a0, a1, bf;
        a0.q[0] = *(const Q128*)(a0_base + kb);
        a0.q[1] = *(const Q128*)(a0_base + kb + 16);
        a1.q[0] = *(const Q128*)(a1_base + kb);
        a1.q[1] = *(const Q128*)(a1_base + kb + 16);
        bf.q[0] = *(const Q128*)(b_base + kb);
        bf.q[1] = *(const Q128*)(b_base + kb + 16);
        acc0 = __builtin_amdgcn_wmma_f32_16x16x32_bf16(false, a0.v, false, bf.v,
                                                       (short)0, acc0, false, false);
        acc1 = __builtin_amdgcn_wmma_f32_16x16x32_bf16(false, a1.v, false, bf.v,
                                                       (short)0, acc1, false, false);
    }

    if (vc < V_) {
        float bias = blin[vc];
#pragma unroll
        for (int i = 0; i < 8; ++i) {
            int r = i + half * 8;   // C/D layout: VGPR i -> row i (lanes 0-15) / row i+8
            logits[(size_t)r * V_ + vc]        = acc0[i] + bias;
            logits[(size_t)(r + 16) * V_ + vc] = acc1[i] + bias;
        }
    }
}

// ---------------- per step: softmax + Gumbel-max sample + embedding gather ---
__global__ __launch_bounds__(512)
void k_softmax_sample(const float* __restrict__ logits, float* __restrict__ out,
                      const float* __restrict__ emb, float* __restrict__ xnext, int t) {
    __shared__ float sA[512];
    __shared__ float sB[512];
    __shared__ int   sI[512];
    const int b   = blockIdx.x;
    const int tid = threadIdx.x;
    const float* lr = logits + (size_t)b * V_;

    float lmax = -3.4e38f, gbest = -3.4e38f;
    int   gbi  = 0;
    unsigned int seed0 = (unsigned int)(t * 1664525 + b * 1013904223 + 12345);
    for (int v = tid; v < V_; v += 512) {
        float l = lr[v];
        lmax = fmaxf(lmax, l);
        unsigned int s = (unsigned int)v * 2654435761u ^ seed0;   // PCG-style hash
        s = s * 747796405u + 2891336453u;
        unsigned int w = ((s >> ((s >> 28) + 4u)) ^ s) * 277803737u;
        w = (w >> 22) ^ w;
        float u = ((float)(w >> 8) + 0.5f) * (1.0f / 16777216.0f);
        float g = -__logf(-__logf(u));                            // Gumbel(0,1)
        float sc = l + g;
        if (sc > gbest) { gbest = sc; gbi = v; }
    }
    sA[tid] = lmax; sB[tid] = gbest; sI[tid] = gbi;
    __syncthreads();
    for (int s = 256; s > 0; s >>= 1) {
        if (tid < s) {
            sA[tid] = fmaxf(sA[tid], sA[tid + s]);
            if (sB[tid + s] > sB[tid]) { sB[tid] = sB[tid + s]; sI[tid] = sI[tid + s]; }
        }
        __syncthreads();
    }
    const float rowmax = sA[0];
    const int   idx    = sI[0];
    __syncthreads();

    float lsum = 0.f;
    for (int v = tid; v < V_; v += 512) lsum += __expf(lr[v] - rowmax);
    sA[tid] = lsum;
    __syncthreads();
    for (int s = 256; s > 0; s >>= 1) {
        if (tid < s) sA[tid] += sA[tid + s];
        __syncthreads();
    }
    const float inv = 1.0f / sA[0];

    float* orow = out + ((size_t)b * T_ + t) * V_;
    for (int v = tid; v < V_; v += 512)
        orow[v] = __expf(lr[v] - rowmax) * inv;

    if (tid < E_) xnext[b * E_ + tid] = emb[(size_t)idx * E_ + tid];
}

extern "C" void kernel_launch(void* const* d_in, const int* in_sizes, int n_in,
                              void* d_out, int out_size, void* d_ws, size_t ws_size,
                              hipStream_t stream) {
    (void)in_sizes; (void)n_in; (void)out_size; (void)ws_size;
    const float* x_in = (const float*)d_in[0];
    // d_in[1] = max_seq_length (scalar, == T_, hardcoded)
    const float* emb  = (const float*)d_in[2];
    const float* Wih  = (const float*)d_in[3];
    const float* Whh  = (const float*)d_in[4];
    const float* bih  = (const float*)d_in[5];
    const float* bhh  = (const float*)d_in[6];
    const float* Wlin = (const float*)d_in[7];
    const float* blin = (const float*)d_in[8];
    float* out = (float*)d_out;

    // carve workspace (256B aligned slices); total ~19.4 MB
    char*  ws  = (char*)d_ws;
    size_t off = 0;
    auto carve = [&](size_t bytes) -> char* {
        char* p = ws + off;
        off += (bytes + 255) & ~(size_t)255;
        return p;
    };
    unsigned short* wbf    = (unsigned short*)carve((size_t)V_ * H_ * 2); // 12.9 MB, L2 resident
    float*          logits = (float*)carve((size_t)B_ * V_ * 4);          // 6.4 MB, L2 resident
    float*          h0     = (float*)carve((size_t)B_ * H_ * 4);
    float*          h1     = (float*)carve((size_t)B_ * H_ * 4);
    float*          cbuf   = (float*)carve((size_t)B_ * H_ * 4);
    unsigned short* hbf    = (unsigned short*)carve((size_t)B_ * H_ * 2);
    float*          xbuf   = (float*)carve((size_t)B_ * E_ * 4);

    hipMemsetAsync(h0,   0, (size_t)B_ * H_ * 4, stream);
    hipMemsetAsync(cbuf, 0, (size_t)B_ * H_ * 4, stream);

    const int n = V_ * H_;
    k_f32_to_bf16<<<(n + 255) / 256, 256, 0, stream>>>(Wlin, wbf, n);

    for (int t = 0; t < T_; ++t) {
        const float* xt    = (t == 0) ? x_in : xbuf;
        float*       hprev = (t & 1) ? h1 : h0;
        float*       hcur  = (t & 1) ? h0 : h1;
        k_lstm_cell<<<(B_ * H_) / 256, 256, 0, stream>>>(xt, hprev, cbuf, hcur, hbf,
                                                         Wih, Whh, bih, bhh);
        k_logits_wmma<<<(V_ + 127) / 128, 256, 0, stream>>>(hbf, wbf, blin, logits);
        k_softmax_sample<<<B_, 512, 0, stream>>>(logits, out, emb, xbuf, t);
    }
}
